// Struct2Vec_506806141452
// MI455X (gfx1250) — matrix operand order repaired and checked by reading, not compiled, not run on hardware
//
#include <hip/hip_runtime.h>

// Struct2Vec message passing for MI455X (gfx1250, wave32, WMMA).
//
// N=100 nodes, B=64 batch, p=128 hidden.
//   item2[n,b,q] = S[n,b] * v2[q]  (relu(D*w3) = D*relu(w3) since D>=0)
//   rounds: mu <- relu((sum_n mu - mu) @ W1^T + const); round 1 = relu(const)
// bf16 WMMA (f16 would overflow: mu grows ~x100/round; bf16 keeps f32 exponent).
// All f32->bf16 conversion is hoisted out of the WMMA kernel:
//   - W1 converted to bf16 once (w1bf)
//   - agg = total - mu converted to bf16 per round (aggbf)
// so the GEMM kernel is pure loads + v_wmma + fused relu epilogue.

typedef __attribute__((ext_vector_type(16))) __bf16        v16bf;
typedef __attribute__((ext_vector_type(8)))  float         v8f;
typedef __attribute__((ext_vector_type(4)))  unsigned int  v4u;
typedef __attribute__((ext_vector_type(8)))  unsigned int  v8u;

#define N_NODES 100
#define BATCH   64
#define P       128

__device__ __forceinline__ unsigned bf16rne(float f) {
  union { float f; unsigned u; } x; x.f = f;
  return (x.u + 0x7FFFu + ((x.u >> 16) & 1u)) >> 16;   // round-to-nearest-even
}
__device__ __forceinline__ unsigned pk_bf16(float lo, float hi) {
  return bf16rne(lo) | (bf16rne(hi) << 16);
}
__device__ __forceinline__ v16bf as_v16bf(v8u u) {
  union { v8u u; v16bf b; } c; c.u = u; return c.b;
}

// v2[q] = sum_p relu(W3[p]) * W2[q,p]       (1 block x 128 threads)
__global__ void s2v_v2_kernel(const float* __restrict__ W3, const float* __restrict__ W2,
                              float* __restrict__ v2) {
  int q = threadIdx.x;
  float acc = 0.f;
  for (int p = 0; p < P; ++p) {
    float w = W3[p];
    w = w > 0.f ? w : 0.f;
    acc += w * W2[q * P + p];
  }
  v2[q] = acc;
}

// W1 (128x128 f32, row-major (q,p)) -> bf16, same layout. One-time, 32 KB.
__global__ void s2v_w1bf_kernel(const float* __restrict__ W1, unsigned* __restrict__ w1bf) {
  int t = blockIdx.x * blockDim.x + threadIdx.x;     // 0..8191, 2 elements each
  w1bf[t] = pk_bf16(W1[2 * t], W1[2 * t + 1]);
}

// S[n,b] = sum_m ||coords[n,b] - coords[m,b]||
__global__ void s2v_dist_kernel(const float* __restrict__ in, float* __restrict__ S) {
  int t = blockIdx.x * blockDim.x + threadIdx.x;
  if (t >= N_NODES * BATCH) return;
  int b = t & (BATCH - 1);
  float x0 = in[t * 6 + 0];
  float y0 = in[t * 6 + 1];
  float s = 0.f;
  for (int m = 0; m < N_NODES; ++m) {
    int mb = m * BATCH + b;
    float dx = x0 - in[mb * 6 + 0];
    float dy = y0 - in[mb * 6 + 1];
    s += sqrtf(dx * dx + dy * dy);
  }
  S[t] = s;
}

// const[n,b,q] = S[n,b]*v2[q] + item3;  mu1 = relu(const)   (grid N*B x 128)
__global__ void s2v_const_kernel(const float* __restrict__ in, const float* __restrict__ S,
                                 const float* __restrict__ v2, const float* __restrict__ W4,
                                 const float* __restrict__ W5, float* __restrict__ cst,
                                 float* __restrict__ mu1) {
  int nb = blockIdx.x;        // n*64 + b
  int q  = threadIdx.x;
  int n  = nb >> 6;
  float c = S[nb] * v2[q];
  if (n == 0) {
    c += in[nb * 6 + 0] * W5[q * 2 + 0] + in[nb * 6 + 1] * W5[q * 2 + 1];
  } else {
    float t = 0.f;
#pragma unroll
    for (int f = 0; f < 6; ++f) t += in[nb * 6 + f] * W4[q * 6 + f];
    c += t;
  }
  cst[nb * P + q] = c;
  float r = c > 0.f ? c : 0.f;
  mu1[nb * P + q] = r;
}

// total[b,q] = sum_n mu[n,b,q]   (8192 threads, coalesced)
__global__ void s2v_total_kernel(const float* __restrict__ mu, float* __restrict__ total) {
  int t = blockIdx.x * blockDim.x + threadIdx.x;   // 0..8191
  float s = 0.f;
  for (int n = 0; n < N_NODES; ++n) s += mu[n * (BATCH * P) + t];
  total[t] = s;
}

// aggbf[r,k] = bf16(total[b,k] - mu[r,k]); 2 elements/thread, packed store.
__global__ void s2v_agg_kernel(const float* __restrict__ mu, const float* __restrict__ total,
                               unsigned* __restrict__ aggbf) {
  int t = blockIdx.x * blockDim.x + threadIdx.x;   // 0..409599
  int e = t * 2;
  int r = e >> 7;                                  // row = n*64+b
  int k = e & (P - 1);
  const float* tr = total + (r & (BATCH - 1)) * P;
  aggbf[t] = pk_bf16(tr[k] - mu[e], tr[k + 1] - mu[e + 1]);
}

// muOut[r,q] = relu( sum_k aggbf[r,k] * w1bf[q,k] + cst[r,q] )
// One wave per 16x16 tile; block = 8 waves = the 8 column tiles of one row tile.
__global__ void s2v_gemm_kernel(const unsigned short* __restrict__ aggbf,
                                const unsigned short* __restrict__ w1bf,
                                const float* __restrict__ cst,
                                float* __restrict__ muOut) {
  const int wave   = threadIdx.x >> 5;   // 0..7 -> column tile
  const int lane   = threadIdx.x & 31;
  const int tile_m = blockIdx.x;         // 0..399
  const int l16    = lane & 15;
  const int hi     = lane >> 4;          // 0/1 half of wave

  const int rowA = tile_m * 16 + l16;
  const unsigned short* __restrict__ aRow = aggbf + rowA * P;

  const int col = wave * 16 + l16;
  const unsigned short* __restrict__ bRow = w1bf + col * P;

  v8f acc = {};
#pragma unroll
  for (int kb = 0; kb < 4; ++kb) {
    // A 16x32 bf16 layout: lanes<16 hold K={kb*32+0..7, +16..23}; lanes>=16: +8 of each.
    const int ka0 = kb * 32 + hi * 8;
    v4u alo = *(const v4u*)(aRow + ka0);          // 8 bf16, 16B aligned
    v4u ahi = *(const v4u*)(aRow + ka0 + 16);     // 8 bf16
    v16bf a = as_v16bf(__builtin_shufflevector(alo, ahi, 0, 1, 2, 3, 4, 5, 6, 7));
    // B 32x16 bf16 layout: same K order as A (lanes<16: +0..7, +16..23; lanes>=16: +8 of each).
    v4u blo = *(const v4u*)(bRow + ka0);
    v4u bhi = *(const v4u*)(bRow + ka0 + 16);
    v16bf b = as_v16bf(__builtin_shufflevector(blo, bhi, 0, 1, 2, 3, 4, 5, 6, 7));

    acc = __builtin_amdgcn_wmma_f32_16x16x32_bf16(
        /*neg_a=*/false, a, /*neg_b=*/false, b,
        /*c_mod=*/(short)0, acc, /*reuse_a=*/false, /*reuse_b=*/false);
  }

  // D layout: VGPR j -> row M=j (lanes 0-15) / M=j+8 (lanes 16-31), col = lane&15.
#pragma unroll
  for (int j = 0; j < 8; ++j) {
    const int row = tile_m * 16 + j + hi * 8;
    const int idx = row * P + col;
    float v = acc[j] + cst[idx];
    muOut[idx] = v > 0.f ? v : 0.f;
  }
}

extern "C" void kernel_launch(void* const* d_in, const int* in_sizes, int n_in,
                              void* d_out, int out_size, void* d_ws, size_t ws_size,
                              hipStream_t stream) {
  (void)in_sizes; (void)n_in; (void)out_size; (void)ws_size;
  const float* in = (const float*)d_in[0];
  const float* W1 = (const float*)d_in[1];
  const float* W2 = (const float*)d_in[2];
  const float* W3 = (const float*)d_in[3];
  const float* W4 = (const float*)d_in[4];
  const float* W5 = (const float*)d_in[5];
  float* out = (float*)d_out;

  // ws layout (floats): v2[128] | S[6400] | total[8192] | const[819200] | muA[819200]
  // then (16-bit): aggbf[819200] | w1bf[16384]   -> ~8.3 MB total
  float* ws  = (float*)d_ws;
  float* v2  = ws;
  float* S   = v2 + P;
  float* tot = S + N_NODES * BATCH;
  float* cst = tot + BATCH * P;
  float* muA = cst + N_NODES * BATCH * P;
  unsigned short* aggbf = (unsigned short*)(muA + N_NODES * BATCH * P);
  unsigned short* w1bf  = aggbf + N_NODES * BATCH * P;

  s2v_v2_kernel<<<1, P, 0, stream>>>(W3, W2, v2);
  s2v_w1bf_kernel<<<(P * P / 2) / 256, 256, 0, stream>>>(W1, (unsigned*)w1bf);
  s2v_dist_kernel<<<(N_NODES * BATCH + 255) / 256, 256, 0, stream>>>(in, S);
  // also produces round-1 result: muA = relu(const)
  s2v_const_kernel<<<N_NODES * BATCH, P, 0, stream>>>(in, S, v2, W4, W5, cst, muA);

  // Rounds 2..4, ping-pong muA <-> d_out so round 4 lands in d_out.
  const float* rin[3]  = { muA, out, muA };
  float*       rout[3] = { out, muA, out };
  const int elems = N_NODES * BATCH * P;
  for (int r = 0; r < 3; ++r) {
    s2v_total_kernel<<<(BATCH * P) / 256, 256, 0, stream>>>(rin[r], tot);
    s2v_agg_kernel<<<(elems / 2) / 256, 256, 0, stream>>>(rin[r], tot, (unsigned*)aggbf);
    s2v_gemm_kernel<<<(N_NODES * BATCH) / 16, 256, 0, stream>>>(aggbf, w1bf, cst, rout[r]);
  }
}
